// GPTLanguageModel_78005196030647
// MI455X (gfx1250) — compile-verified, matrix operand order checked by reading, not run on hardware
//
#include <hip/hip_runtime.h>
#include <math.h>

typedef __attribute__((ext_vector_type(16))) _Float16 v16h;
typedef __attribute__((ext_vector_type(8)))  _Float16 v8h;
typedef __attribute__((ext_vector_type(8)))  float    v8f;
typedef __attribute__((ext_vector_type(4)))  unsigned int v4u;
typedef __attribute__((ext_vector_type(8)))  int v8i;
typedef __attribute__((ext_vector_type(4)))  int v4i;

#if __has_builtin(__builtin_amdgcn_tensor_load_to_lds) && __has_builtin(__builtin_amdgcn_s_wait_tensorcnt)
#define HAVE_TDM 1
#else
#define HAVE_TDM 0
#endif

// ---------------- problem constants ----------------
namespace {
constexpr int L   = 12;
constexpr int D   = 726;
constexpr int H   = 16;
constexpr int HD  = 45;
constexpr int V   = 50257;
constexpr int Bn  = 16;
constexpr int T   = 258;
constexpr int FF  = 4 * D;          // 2904
constexpr int NT  = Bn * T;         // 4128 tokens = 129*32
constexpr int NTP = 4160;           // tokens padded to mult of 64 (A rows, big GEMMs)
constexpr int DP  = 736;            // D   padded to mult of 32 (K dim)
constexpr int DPN = 768;            // D   padded to mult of 64 (B^T rows)
constexpr int FFP = 2912;           // FF  padded to mult of 32 (K dim)
constexpr int FFN = 2944;           // FF  padded to mult of 64 (B^T rows)
constexpr int HDK = 64;             // HD  padded as K-dim
constexpr int TP  = 288;            // T   padded to mult of 32 (rows and K)
constexpr int VN  = 50304;          // V   padded to mult of 64 (B^T rows)
constexpr int NQ  = 64;             // HD  padded to mult of 32 (B^T rows, per-head)
constexpr float ATT_SCALE = 0.14907119849998599f;  // 1/sqrt(45)
}

// ---------------- WMMA fragment load ----------------
// A fragment (16x32 f16): lane q=lane&15 -> row, half hf=lane>>4.
// v16h[0..7]  = row[k + hf*8 + 0..7]
// v16h[8..15] = row[k + 16 + hf*8 + 0..7]
// B fragment identical, loaded from B^T (row-major [N][K]).
__device__ __forceinline__ v16h load_frag(const _Float16* rowp, int kbase, int hf) {
  const _Float16* p = rowp + kbase + hf * 8;
  v8h lo = *(const v8h*)(p);
  v8h hi = *(const v8h*)(p + 16);
  v16h r;
#pragma unroll
  for (int i = 0; i < 8; ++i) { r[i] = lo[i]; r[i + 8] = hi[i]; }
  return r;
}

// 2x2 blocking: one wave computes a 32x32 tile (attention-side GEMMs)
__device__ __forceinline__ void wmma_tile22(const _Float16* __restrict__ A, int lda,
                                            const _Float16* __restrict__ BT, int ldb,
                                            int m0, int n0, int K, v8f acc[2][2]) {
  const int lane = threadIdx.x & 31;
  const int q  = lane & 15;
  const int hf = lane >> 4;
  const _Float16* ar0 = A  + (size_t)(m0 + q) * lda;
  const _Float16* ar1 = ar0 + (size_t)16 * lda;
  const _Float16* br0 = BT + (size_t)(n0 + q) * ldb;
  const _Float16* br1 = br0 + (size_t)16 * ldb;
#pragma unroll
  for (int i = 0; i < 2; ++i)
#pragma unroll
    for (int j = 0; j < 2; ++j)
      acc[i][j] = (v8f){0.f, 0.f, 0.f, 0.f, 0.f, 0.f, 0.f, 0.f};
  for (int k = 0; k < K; k += 32) {
    v16h a0 = load_frag(ar0, k, hf);
    v16h a1 = load_frag(ar1, k, hf);
    v16h b0 = load_frag(br0, k, hf);
    v16h b1 = load_frag(br1, k, hf);
    acc[0][0] = __builtin_amdgcn_wmma_f32_16x16x32_f16(false, a0, false, b0, (short)0, acc[0][0], false, false);
    acc[0][1] = __builtin_amdgcn_wmma_f32_16x16x32_f16(false, a0, false, b1, (short)0, acc[0][1], false, false);
    acc[1][0] = __builtin_amdgcn_wmma_f32_16x16x32_f16(false, a1, false, b0, (short)0, acc[1][0], false, false);
    acc[1][1] = __builtin_amdgcn_wmma_f32_16x16x32_f16(false, a1, false, b1, (short)0, acc[1][1], false, false);
  }
}

#if HAVE_TDM
// Issue one TDM 2D-tile load: rows x cols f16 tile from global (row stride ldbytes/2)
// into LDS at lds_byte. D# layout per CDNA5 ISA.
__device__ __forceinline__ void tdm_load_tile(const _Float16* gsrc, int ldb,
                                              unsigned lds_byte, int rows, int cols) {
  unsigned long long ga = (unsigned long long)(const void*)gsrc;
  v4u g0;
  g0[0] = 1u;                                        // count=1, is_restore=0, gather off
  g0[1] = lds_byte;                                  // lds_addr (bytes)
  g0[2] = (unsigned)ga;                              // global_addr[31:0]
  g0[3] = (unsigned)((ga >> 32) & 0x01FFFFFFull) | (2u << 30);  // addr[56:32] | type=2
  v8i g1;
  g1[0] = (1 << 16);                                 // data_size=1 (2 bytes)
  g1[1] = (cols & 0xFFFF) << 16;                     // tensor_dim0[15:0]
  g1[2] = (rows & 0xFFFF) << 16;                     // tensor_dim0 hi=0 | tensor_dim1[15:0]
  g1[3] = (cols & 0xFFFF) << 16;                     // tensor_dim1 hi=0 | tile_dim0
  g1[4] = rows & 0xFFFF;                             // tile_dim1 | tile_dim2=0
  g1[5] = ldb;                                       // tensor_dim0_stride[31:0]
  g1[6] = 0;
  g1[7] = 0;
  v4i z4 = (v4i){0, 0, 0, 0};
#if __clang_major__ >= 23
  v8i z8 = (v8i){0, 0, 0, 0, 0, 0, 0, 0};
  __builtin_amdgcn_tensor_load_to_lds(g0, g1, z4, z4, z8, 0);
#else
  __builtin_amdgcn_tensor_load_to_lds(g0, g1, z4, z4, 0);
#endif
}

// 4x4 blocking with TDM-staged B tiles (double buffered in LDS).
// One wave = 64x64 tile; B chunk (64 rows x 32 k, 4KB) streamed by the TDM
// while the wave runs 16 WMMAs on the previous chunk.
__device__ __forceinline__ void wmma_tile44_tdm(const _Float16* __restrict__ A, int lda,
                                                const _Float16* __restrict__ BT, int ldb,
                                                int m0, int n0, int K,
                                                _Float16 (*bs)[64][32], v8f acc[4][4]) {
  const int lane = threadIdx.x & 31;
  const int q  = lane & 15;
  const int hf = lane >> 4;
  const _Float16* ar[4];
#pragma unroll
  for (int i = 0; i < 4; ++i)
    ar[i] = A + (size_t)(m0 + 16 * i + q) * lda;
#pragma unroll
  for (int i = 0; i < 4; ++i)
#pragma unroll
    for (int j = 0; j < 4; ++j)
      acc[i][j] = (v8f){0.f, 0.f, 0.f, 0.f, 0.f, 0.f, 0.f, 0.f};

  const _Float16* bbase = BT + (size_t)n0 * ldb;
  const int NC = K / 32;
  tdm_load_tile(bbase, ldb, (unsigned)(unsigned long long)(const void*)&bs[0][0][0], 64, 32);
  for (int c = 0; c < NC; ++c) {
    if (c + 1 < NC) {
      tdm_load_tile(bbase + (c + 1) * 32, ldb,
                    (unsigned)(unsigned long long)(const void*)&bs[(c + 1) & 1][0][0], 64, 32);
      __builtin_amdgcn_s_wait_tensorcnt(1);  // chunk c complete (in-order per wave)
    } else {
      __builtin_amdgcn_s_wait_tensorcnt(0);
    }
    const _Float16* bsrc = &bs[c & 1][0][0];
    v16h a[4], b[4];
#pragma unroll
    for (int i = 0; i < 4; ++i) a[i] = load_frag(ar[i], c * 32, hf);
#pragma unroll
    for (int j = 0; j < 4; ++j) b[j] = load_frag(bsrc + (size_t)(q + 16 * j) * 32, 0, hf);
#pragma unroll
    for (int i = 0; i < 4; ++i)
#pragma unroll
      for (int j = 0; j < 4; ++j)
        acc[i][j] = __builtin_amdgcn_wmma_f32_16x16x32_f16(false, a[i], false, b[j],
                                                           (short)0, acc[i][j], false, false);
  }
}
#else
// fallback: 4x4 blocking with direct global B loads
__device__ __forceinline__ void wmma_tile44_dir(const _Float16* __restrict__ A, int lda,
                                                const _Float16* __restrict__ BT, int ldb,
                                                int m0, int n0, int K, v8f acc[4][4]) {
  const int lane = threadIdx.x & 31;
  const int q  = lane & 15;
  const int hf = lane >> 4;
  const _Float16* ar[4];
  const _Float16* br[4];
#pragma unroll
  for (int i = 0; i < 4; ++i) {
    ar[i] = A  + (size_t)(m0 + 16 * i + q) * lda;
    br[i] = BT + (size_t)(n0 + 16 * i + q) * ldb;
  }
#pragma unroll
  for (int i = 0; i < 4; ++i)
#pragma unroll
    for (int j = 0; j < 4; ++j)
      acc[i][j] = (v8f){0.f, 0.f, 0.f, 0.f, 0.f, 0.f, 0.f, 0.f};
  for (int k = 0; k < K; k += 32) {
    v16h a[4], b[4];
#pragma unroll
    for (int i = 0; i < 4; ++i) a[i] = load_frag(ar[i], k, hf);
#pragma unroll
    for (int j = 0; j < 4; ++j) b[j] = load_frag(br[j], k, hf);
#pragma unroll
    for (int i = 0; i < 4; ++i)
#pragma unroll
      for (int j = 0; j < 4; ++j)
        acc[i][j] = __builtin_amdgcn_wmma_f32_16x16x32_f16(false, a[i], false, b[j],
                                                           (short)0, acc[i][j], false, false);
  }
}
#endif
// C/D layout: element r of acc[i][j]: m = m0 + i*16 + r + 8*(lane>>4), n = n0 + j*16 + (lane&15)

// ---------------- misc kernels ----------------
__global__ void k_embed(const int* __restrict__ idx, const float* __restrict__ tok,
                        const float* __restrict__ pos, float* __restrict__ x) {
  size_t i = (size_t)blockIdx.x * 256 + threadIdx.x;
  if (i >= (size_t)NT * D) return;
  int bt = (int)(i / D), d = (int)(i % D);
  int t  = bt % T;
  x[i] = tok[(size_t)idx[bt] * D + d] + pos[(size_t)t * D + d];
}

// transpose + convert weights: out[n][k] = (f16) in[k][n], zero for k>=K
__global__ void k_wconv(const float* __restrict__ in, _Float16* __restrict__ out,
                        int K, int N, int KP, long long inStride, long long outStride) {
  long long i = (long long)blockIdx.x * 256 + threadIdx.x;
  if (i >= (long long)N * KP) return;
  int n = (int)(i / KP), k = (int)(i % KP);
  const float* inm = in + (long long)blockIdx.y * inStride;
  _Float16* outm   = out + (long long)blockIdx.y * outStride;
  outm[(size_t)n * KP + k] = (k < K) ? (_Float16)inm[(size_t)k * N + n] : (_Float16)0.f;
}

// layernorm over D=726, output f16 padded to DP with zeros
__global__ void k_ln(const float* __restrict__ x, const float* __restrict__ g,
                     const float* __restrict__ bt, _Float16* __restrict__ out) {
  __shared__ float sm[256];
  const int row = blockIdx.x, tid = threadIdx.x;
  const float* xr = x + (size_t)row * D;
  float s = 0.f, s2 = 0.f;
  for (int d = tid; d < D; d += 256) { float v = xr[d]; s += v; s2 += v * v; }
  sm[tid] = s; __syncthreads();
  for (int st = 128; st > 0; st >>= 1) { if (tid < st) sm[tid] += sm[tid + st]; __syncthreads(); }
  float mean = sm[0] / D; __syncthreads();
  sm[tid] = s2; __syncthreads();
  for (int st = 128; st > 0; st >>= 1) { if (tid < st) sm[tid] += sm[tid + st]; __syncthreads(); }
  float var  = sm[0] / D - mean * mean;
  float rstd = rsqrtf(var + 1e-5f);
  _Float16* orow = out + (size_t)row * DP;
  for (int d = tid; d < DP; d += 256)
    orow[d] = (d < D) ? (_Float16)((xr[d] - mean) * rstd * g[d] + bt[d]) : (_Float16)0.f;
}

// causal softmax over row t (cols 0..t), f16 out padded [TP][TP]
__global__ void k_softmax(const float* __restrict__ attf, _Float16* __restrict__ attp) {
  __shared__ float sm[256];
  const int t = blockIdx.x, bh = blockIdx.y, tid = threadIdx.x;
  const float* row = attf + ((size_t)bh * T + t) * T;
  float mx = -INFINITY;
  for (int s = tid; s <= t; s += 256) mx = fmaxf(mx, row[s]);
  sm[tid] = mx; __syncthreads();
  for (int st = 128; st > 0; st >>= 1) { if (tid < st) sm[tid] = fmaxf(sm[tid], sm[tid + st]); __syncthreads(); }
  mx = sm[0]; __syncthreads();
  float sum = 0.f;
  for (int s = tid; s <= t; s += 256) sum += __expf(row[s] - mx);
  sm[tid] = sum; __syncthreads();
  for (int st = 128; st > 0; st >>= 1) { if (tid < st) sm[tid] += sm[tid + st]; __syncthreads(); }
  float inv = 1.f / sm[0];
  _Float16* orow = attp + ((size_t)bh * TP + t) * TP;
  for (int s = tid; s < T; s += 256)
    orow[s] = (s <= t) ? (_Float16)(__expf(row[s] - mx) * inv) : (_Float16)0.f;
}

// ---------------- attention-side GEMMs (1 wave = 32x32 tile) ----------------
// q/k: out[b][h][t][HDK]; v (vmode=1): out[b][h][n][TP] (transposed)
__global__ void k_gemm_qkv(const _Float16* __restrict__ A, const _Float16* __restrict__ WT,
                           _Float16* __restrict__ out, int vmode) {
  const int n0 = blockIdx.x * 32, m0 = blockIdx.y * 32, hh = blockIdx.z;
  const _Float16* BT = WT + (size_t)hh * NQ * DP;
  v8f acc[2][2];
  wmma_tile22(A, DP, BT, DP, m0, n0, DP, acc);
  const int lane = threadIdx.x & 31, q = lane & 15, hf = lane >> 4;
#pragma unroll
  for (int i = 0; i < 2; ++i)
#pragma unroll
    for (int j = 0; j < 2; ++j)
#pragma unroll
      for (int r = 0; r < 8; ++r) {
        int m = m0 + i * 16 + r + 8 * hf, n = n0 + j * 16 + q;
        if (n >= HD) continue;
        int b = m / T, t = m - b * T;
        if (vmode)
          out[(((size_t)b * H + hh) * NQ + n) * TP + t] = (_Float16)acc[i][j][r];
        else
          out[(((size_t)b * H + hh) * TP + t) * HDK + n] = (_Float16)acc[i][j][r];
      }
}

__global__ void k_gemm_att(const _Float16* __restrict__ qb, const _Float16* __restrict__ kb,
                           float* __restrict__ attf) {
  const int n0 = blockIdx.x * 32, m0 = blockIdx.y * 32, bh = blockIdx.z;
  const _Float16* A  = qb + (size_t)bh * TP * HDK;
  const _Float16* BT = kb + (size_t)bh * TP * HDK;
  v8f acc[2][2];
  wmma_tile22(A, HDK, BT, HDK, m0, n0, HDK, acc);
  const int lane = threadIdx.x & 31, q = lane & 15, hf = lane >> 4;
#pragma unroll
  for (int i = 0; i < 2; ++i)
#pragma unroll
    for (int j = 0; j < 2; ++j)
#pragma unroll
      for (int r = 0; r < 8; ++r) {
        int m = m0 + i * 16 + r + 8 * hf, n = n0 + j * 16 + q;
        if (m < T && n < T)
          attf[((size_t)bh * T + m) * T + n] = acc[i][j][r] * ATT_SCALE;
      }
}

__global__ void k_gemm_o(const _Float16* __restrict__ attp, const _Float16* __restrict__ vt,
                         _Float16* __restrict__ obuf) {
  const int n0 = blockIdx.x * 32, m0 = blockIdx.y * 32, bh = blockIdx.z;
  const _Float16* A  = attp + (size_t)bh * TP * TP;
  const _Float16* BT = vt   + (size_t)bh * NQ * TP;
  v8f acc[2][2];
  wmma_tile22(A, TP, BT, TP, m0, n0, TP, acc);
  const int lane = threadIdx.x & 31, q = lane & 15, hf = lane >> 4;
  const int b = bh / H, hh = bh % H;
#pragma unroll
  for (int i = 0; i < 2; ++i)
#pragma unroll
    for (int j = 0; j < 2; ++j)
#pragma unroll
      for (int r = 0; r < 8; ++r) {
        int m = m0 + i * 16 + r + 8 * hf, n = n0 + j * 16 + q;
        if (m < T && n < HD)
          obuf[((size_t)b * T + m) * DP + hh * HD + n] = (_Float16)acc[i][j][r];
      }
}

// ---------------- large token GEMMs (1 wave = 64x64 tile, TDM-staged B) ----------------
// fp32 output + bias, optional in-place residual add (proj / ff2 / lm-head)
__global__ void k_gemm_f32(const _Float16* __restrict__ A, int lda,
                           const _Float16* __restrict__ BT, int ldb, int K,
                           float* __restrict__ C, int ldc, int Nreal,
                           const float* __restrict__ bias, int addResid) {
  const int n0 = blockIdx.x * 64, m0 = blockIdx.y * 64;
  v8f acc[4][4];
#if HAVE_TDM
  __shared__ _Float16 bs[2][64][32];
  wmma_tile44_tdm(A, lda, BT, ldb, m0, n0, K, bs, acc);
#else
  wmma_tile44_dir(A, lda, BT, ldb, m0, n0, K, acc);
#endif
  const int lane = threadIdx.x & 31, q = lane & 15, hf = lane >> 4;
#pragma unroll
  for (int i = 0; i < 4; ++i)
#pragma unroll
    for (int j = 0; j < 4; ++j)
#pragma unroll
      for (int r = 0; r < 8; ++r) {
        int m = m0 + i * 16 + r + 8 * hf, n = n0 + j * 16 + q;
        if (m >= NT || n >= Nreal) continue;
        size_t o = (size_t)m * ldc + n;
        float v = acc[i][j][r] + bias[n];
        if (addResid) v += C[o];
        C[o] = v;
      }
}

// f16 output with bias + ReLU (ff1)
__global__ void k_gemm_ff1(const _Float16* __restrict__ A, const _Float16* __restrict__ BT,
                           const float* __restrict__ bias, _Float16* __restrict__ out) {
  const int n0 = blockIdx.x * 64, m0 = blockIdx.y * 64;
  v8f acc[4][4];
#if HAVE_TDM
  __shared__ _Float16 bs[2][64][32];
  wmma_tile44_tdm(A, DP, BT, DP, m0, n0, DP, bs, acc);
#else
  wmma_tile44_dir(A, DP, BT, DP, m0, n0, DP, acc);
#endif
  const int lane = threadIdx.x & 31, q = lane & 15, hf = lane >> 4;
#pragma unroll
  for (int i = 0; i < 4; ++i)
#pragma unroll
    for (int j = 0; j < 4; ++j)
#pragma unroll
      for (int r = 0; r < 8; ++r) {
        int m = m0 + i * 16 + r + 8 * hf, n = n0 + j * 16 + q;
        if (m < NT && n < FF)
          out[(size_t)m * FFP + n] = (_Float16)fmaxf(acc[i][j][r] + bias[n], 0.f);
      }
}

// ---------------- host ----------------
extern "C" void kernel_launch(void* const* d_in, const int* in_sizes, int n_in,
                              void* d_out, int out_size, void* d_ws, size_t ws_size,
                              hipStream_t stream) {
  (void)in_sizes; (void)n_in; (void)out_size;
  const int*   idx   = (const int*)  d_in[0];
  const float* tok   = (const float*)d_in[1];
  const float* pos   = (const float*)d_in[2];
  const float* Wq    = (const float*)d_in[3];
  const float* Wk    = (const float*)d_in[4];
  const float* Wv    = (const float*)d_in[5];
  const float* Wproj = (const float*)d_in[6];
  const float* bproj = (const float*)d_in[7];
  const float* ln1g  = (const float*)d_in[8];
  const float* ln1b  = (const float*)d_in[9];
  const float* ln2g  = (const float*)d_in[10];
  const float* ln2b  = (const float*)d_in[11];
  const float* W1    = (const float*)d_in[12];
  const float* b1    = (const float*)d_in[13];
  const float* W2    = (const float*)d_in[14];
  const float* b2    = (const float*)d_in[15];
  const float* lnfg  = (const float*)d_in[16];
  const float* lnfb  = (const float*)d_in[17];
  const float* Wlm   = (const float*)d_in[18];
  const float* blm   = (const float*)d_in[19];
  float* out = (float*)d_out;

  // workspace bump allocator (256B aligned)
  char* ws = (char*)d_ws;
  size_t off = 0;
  auto alloc = [&](size_t bytes) -> char* {
    char* p = ws + off;
    off += (bytes + 255) & ~(size_t)255;
    return p;
  };
  _Float16* h16  = (_Float16*)alloc((size_t)NTP * DP * 2);   // padded token rows
  _Float16* qb   = (_Float16*)alloc((size_t)Bn * H * TP * HDK * 2);
  _Float16* kbuf = (_Float16*)alloc((size_t)Bn * H * TP * HDK * 2);
  _Float16* vt   = (_Float16*)alloc((size_t)Bn * H * NQ * TP * 2);
  float*    attf = (float*)   alloc((size_t)Bn * H * T * T * 4);
  _Float16* attp = (_Float16*)alloc((size_t)Bn * H * TP * TP * 2);
  _Float16* obuf = (_Float16*)alloc((size_t)NTP * DP * 2);
  _Float16* ff1b = (_Float16*)alloc((size_t)NTP * FFP * 2);
  float*    xbuf = (float*)   alloc((size_t)NT * D * 4);
  _Float16* WqT  = (_Float16*)alloc((size_t)L * H * NQ * DP * 2);
  _Float16* WkT  = (_Float16*)alloc((size_t)L * H * NQ * DP * 2);
  _Float16* WvT  = (_Float16*)alloc((size_t)L * H * NQ * DP * 2);
  _Float16* WpT  = (_Float16*)alloc((size_t)L * DPN * DP * 2);
  _Float16* W1T  = (_Float16*)alloc((size_t)L * FFN * DP * 2);
  _Float16* W2T  = (_Float16*)alloc((size_t)L * DPN * FFP * 2);
  _Float16* WlmT = (_Float16*)alloc((size_t)VN * DP * 2);
  size_t total = off;

  // zero workspace: establishes all K/M/N padding as zeros
  hipMemsetAsync(d_ws, 0, total <= ws_size ? total : ws_size, stream);

  // ---- weight conversion (f32 -> f16, transposed [N][Kpad]) ----
  {
    long long nel = (long long)HD * DP;
    dim3 g((unsigned)((nel + 255) / 256), L * H);
    k_wconv<<<g, 256, 0, stream>>>(Wq, WqT, D, HD, DP, (long long)D * HD, (long long)NQ * DP);
    k_wconv<<<g, 256, 0, stream>>>(Wk, WkT, D, HD, DP, (long long)D * HD, (long long)NQ * DP);
    k_wconv<<<g, 256, 0, stream>>>(Wv, WvT, D, HD, DP, (long long)D * HD, (long long)NQ * DP);
  }
  {
    long long nel = (long long)D * DP;
    dim3 g((unsigned)((nel + 255) / 256), L);
    k_wconv<<<g, 256, 0, stream>>>(Wproj, WpT, H * HD, D, DP,
                                   (long long)(H * HD) * D, (long long)DPN * DP);
  }
  {
    long long nel = (long long)FF * DP;
    dim3 g((unsigned)((nel + 255) / 256), L);
    k_wconv<<<g, 256, 0, stream>>>(W1, W1T, D, FF, DP, (long long)D * FF, (long long)FFN * DP);
  }
  {
    long long nel = (long long)D * FFP;
    dim3 g((unsigned)((nel + 255) / 256), L);
    k_wconv<<<g, 256, 0, stream>>>(W2, W2T, FF, D, FFP, (long long)FF * D, (long long)DPN * FFP);
  }
  {
    long long nel = (long long)V * DP;
    dim3 g((unsigned)((nel + 255) / 256), 1);
    k_wconv<<<g, 256, 0, stream>>>(Wlm, WlmT, D, V, DP, 0, 0);
  }

  // ---- embedding ----
  {
    long long nel = (long long)NT * D;
    k_embed<<<dim3((unsigned)((nel + 255) / 256)), 256, 0, stream>>>(idx, tok, pos, xbuf);
  }

  // ---- transformer blocks ----
  const int MT32 = NT / 32;   // 129 tiles for 32-row kernels
  const int MT64 = NTP / 64;  // 65 tiles for 64-row kernels
  for (int l = 0; l < L; ++l) {
    k_ln<<<NT, 256, 0, stream>>>(xbuf, ln1g + (size_t)l * D, ln1b + (size_t)l * D, h16);

    dim3 gq(NQ / 32, MT32, H);  // 2 x 129 x 16
    const size_t wqs = (size_t)l * H * NQ * DP;
    k_gemm_qkv<<<gq, 32, 0, stream>>>(h16, WqT + wqs, qb, 0);
    k_gemm_qkv<<<gq, 32, 0, stream>>>(h16, WkT + wqs, kbuf, 0);
    k_gemm_qkv<<<gq, 32, 0, stream>>>(h16, WvT + wqs, vt, 1);

    dim3 ga(TP / 32, TP / 32, Bn * H);  // 9 x 9 x 256
    k_gemm_att<<<ga, 32, 0, stream>>>(qb, kbuf, attf);

    k_softmax<<<dim3(T, Bn * H), 256, 0, stream>>>(attf, attp);

    dim3 go(NQ / 32, TP / 32, Bn * H);  // 2 x 9 x 256
    k_gemm_o<<<go, 32, 0, stream>>>(attp, vt, obuf);

    dim3 gp(DPN / 64, MT64);  // 12 x 65
    k_gemm_f32<<<gp, 32, 0, stream>>>(obuf, DP, WpT + (size_t)l * DPN * DP, DP, DP,
                                      xbuf, D, D, bproj + (size_t)l * D, 1);

    k_ln<<<NT, 256, 0, stream>>>(xbuf, ln2g + (size_t)l * D, ln2b + (size_t)l * D, h16);

    dim3 gf1(FFN / 64, MT64);  // 46 x 65
    k_gemm_ff1<<<gf1, 32, 0, stream>>>(h16, W1T + (size_t)l * FFN * DP,
                                       b1 + (size_t)l * FF, ff1b);

    dim3 gf2(DPN / 64, MT64);  // 12 x 65
    k_gemm_f32<<<gf2, 32, 0, stream>>>(ff1b, FFP, W2T + (size_t)l * DPN * FFP, FFP, FFP,
                                       xbuf, D, D, b2 + (size_t)l * D, 1);
  }

  // ---- final LN + LM head ----
  k_ln<<<NT, 256, 0, stream>>>(xbuf, lnfg, lnfb, h16);
  dim3 glm(VN / 64, MT64);  // 786 x 65
  k_gemm_f32<<<glm, 32, 0, stream>>>(h16, DP, WlmT, DP, DP, out, V, V, blm, 0);
}
